// GPSLayer_10453950398933
// MI455X (gfx1250) — matrix operand order repaired and checked by reading, not compiled, not run on hardware
//
#include <hip/hip_runtime.h>
#include <hip/hip_bf16.h>

// ---------------- problem constants (match reference) ----------------
constexpr int NN  = 4096;     // nodes
constexpr int DD  = 512;      // model dim
constexpr int HH  = 8;        // heads
constexpr int DHH = 64;       // head dim
constexpr int EE  = 131072;   // edges
constexpr float EPS = 1e-5f;

typedef __attribute__((ext_vector_type(16))) __bf16 bf16x16;
typedef __attribute__((ext_vector_type(8)))  __bf16 bf16x8;
typedef __attribute__((ext_vector_type(8)))  float  floatx8;

__device__ __forceinline__ __bf16 f2bf(float f) { return (__bf16)f; }

__device__ __forceinline__ floatx8 zero8() {
  floatx8 z;
#pragma unroll
  for (int i = 0; i < 8; ++i) z[i] = 0.f;
  return z;
}

// ---------------- packed f32 -> bf16 conversion (8 elems / thread) ----------
__global__ void cvt_bf16_kernel(const float* __restrict__ in, __bf16* __restrict__ out, int n) {
  int i = (blockIdx.x * 256 + threadIdx.x) * 8;
  if (i >= n) return;
  float4 v0 = *(const float4*)(in + i);
  float4 v1 = *(const float4*)(in + i + 4);
  bf16x8 r;
  r[0] = f2bf(v0.x); r[1] = f2bf(v0.y); r[2] = f2bf(v0.z); r[3] = f2bf(v0.w);
  r[4] = f2bf(v1.x); r[5] = f2bf(v1.y); r[6] = f2bf(v1.z); r[7] = f2bf(v1.w);
  *(bf16x8*)(out + i) = r;
}

// ------- V transpose: vt[h*64+d][n] = bf16(qkv_f32[n][2D + h*64 + d]) -------
__global__ void v_transpose_kernel(const float* __restrict__ qkvf, __bf16* __restrict__ vt) {
  int hd = blockIdx.x;                           // 0..DD-1 (h*64+d)
  const float* src = qkvf + 2 * DD + hd;
  __bf16* dst = vt + (size_t)hd * NN;
  for (int n = threadIdx.x; n < NN; n += 256)
    dst[n] = f2bf(src[(size_t)n * (3 * DD)]);
}

// ---------------- edge scatter: agg[dst] += relu(x[src] + e) ----------------
__global__ void edge_agg_kernel(const float* __restrict__ x,
                                const int*   __restrict__ ei,
                                const float* __restrict__ ea,
                                float*       __restrict__ agg) {
  int e = blockIdx.x;
  int s = ei[e];
  int d = ei[EE + e];
  int t = threadIdx.x;                               // 0..127 -> 512 floats
  const float4 xv = ((const float4*)(x  + (size_t)s * DD))[t];
  const float4 ev = ((const float4*)(ea + (size_t)e * DD))[t];
  float* ag = agg + (size_t)d * DD + t * 4;
  unsafeAtomicAdd(ag + 0, fmaxf(xv.x + ev.x, 0.f));
  unsafeAtomicAdd(ag + 1, fmaxf(xv.y + ev.y, 0.f));
  unsafeAtomicAdd(ag + 2, fmaxf(xv.z + ev.z, 0.f));
  unsafeAtomicAdd(ag + 3, fmaxf(xv.w + ev.w, 0.f));
}

// ---------------- a += b ----------------
__global__ void add_inplace_kernel(float* __restrict__ a, const float* __restrict__ b, int n) {
  int i = blockIdx.x * 256 + threadIdx.x;
  if (i < n) a[i] += b[i];
}

// ------- WMMA GEMM (bf16 in, f32 out): C = [relu](A @ W^T + bias (+R)) ------
// one wave per block; wave computes a 32x64 tile (2 M-tiles x 4 N-tiles),
// each W fragment feeds two WMMAs.
__global__ __launch_bounds__(32)
void wmma_gemm_kernel(const __bf16* __restrict__ A, const __bf16* __restrict__ W,
                      const float* __restrict__ bias, const float* __restrict__ R,
                      float* __restrict__ C, int M, int Nout, int K, int doRelu) {
  const int lane = threadIdx.x;
  const int lm = lane & 15, half = lane >> 4;
  const int nbase = blockIdx.x * 64;
  const int mbase = blockIdx.y * 32;

  floatx8 acc[2][4];
#pragma unroll
  for (int mt = 0; mt < 2; ++mt)
#pragma unroll
    for (int g = 0; g < 4; ++g) acc[mt][g] = zero8();

  const __bf16* arow0 = A + (size_t)(mbase + lm) * K;
  const __bf16* arow1 = A + (size_t)(mbase + 16 + lm) * K;

  for (int kb = 0; kb < K; kb += 32) {
    bf16x16 a0, a1;
    {
      bf16x8 lo0 = *(const bf16x8*)(arow0 + kb + half * 16);
      bf16x8 hi0 = *(const bf16x8*)(arow0 + kb + half * 16 + 8);
      bf16x8 lo1 = *(const bf16x8*)(arow1 + kb + half * 16);
      bf16x8 hi1 = *(const bf16x8*)(arow1 + kb + half * 16 + 8);
#pragma unroll
      for (int i = 0; i < 8; ++i) {
        a0[i] = lo0[i]; a0[i + 8] = hi0[i];
        a1[i] = lo1[i]; a1[i + 8] = hi1[i];
      }
    }
#pragma unroll
    for (int g = 0; g < 4; ++g) {
      bf16x16 b = *(const bf16x16*)(W + (size_t)(nbase + g * 16 + lm) * K + kb + half * 16);
      acc[0][g] = __builtin_amdgcn_wmma_f32_16x16x32_bf16(
          false, a0, false, b, (short)0, acc[0][g], false, false);
      acc[1][g] = __builtin_amdgcn_wmma_f32_16x16x32_bf16(
          false, a1, false, b, (short)0, acc[1][g], false, false);
    }
  }
#pragma unroll
  for (int mt = 0; mt < 2; ++mt) {
#pragma unroll
    for (int g = 0; g < 4; ++g) {
      int col = nbase + g * 16 + lm;
      float bv = bias ? bias[col] : 0.f;
#pragma unroll
      for (int v = 0; v < 8; ++v) {
        int row = mbase + mt * 16 + v + half * 8;
        float val = acc[mt][g][v] + bv;
        if (R) val += R[(size_t)row * Nout + col];
        if (doRelu) val = fmaxf(val, 0.f);
        C[(size_t)row * Nout + col] = val;
      }
    }
  }
}

// ---------------- flash attention: one wave per (head, 32-query tile) --------
// qkv (bf16, [N,3D]) supplies Q/K row-major; vt (bf16, [D][N]) supplies V^T.
// Exact softmax without max-shift (logits bounded far below exp overflow);
// row-sum kept as per-lane partials, reduced across lanes once at the end.
__global__ __launch_bounds__(32)
void attn_flash_kernel(const __bf16* __restrict__ qkv, const __bf16* __restrict__ vt,
                       float* __restrict__ o) {
  __shared__ __bf16 pS[2][16 * 32];
  const int lane = threadIdx.x;
  const int lm = lane & 15, half = lane >> 4;
  const int qb = blockIdx.x * 32;
  const int h  = blockIdx.y;
  const int LDQ = 3 * DD;
  const float scale = 0.125f;  // 1/sqrt(64)

  // Q fragments: 2 query tiles x (dims 0..31 | 32..63)
  bf16x16 aq[2][2];
#pragma unroll
  for (int qt = 0; qt < 2; ++qt) {
    const __bf16* qrow = qkv + (size_t)(qb + qt * 16 + lm) * LDQ + h * DHH;
    bf16x8 q0 = *(const bf16x8*)(qrow + half * 16);
    bf16x8 q1 = *(const bf16x8*)(qrow + half * 16 + 8);
    bf16x8 q2 = *(const bf16x8*)(qrow + 32 + half * 16);
    bf16x8 q3 = *(const bf16x8*)(qrow + 40 + half * 16);
#pragma unroll
    for (int i = 0; i < 8; ++i) {
      aq[qt][0][i] = q0[i]; aq[qt][0][i + 8] = q1[i];
      aq[qt][1][i] = q2[i]; aq[qt][1][i + 8] = q3[i];
    }
  }

  float Lpart[2][8];
  floatx8 oacc[2][4];
#pragma unroll
  for (int qt = 0; qt < 2; ++qt) {
#pragma unroll
    for (int v = 0; v < 8; ++v) Lpart[qt][v] = 0.f;
#pragma unroll
    for (int g = 0; g < 4; ++g) oacc[qt][g] = zero8();
  }

  for (int kb = 0; kb < NN; kb += 32) {
    // ---- K fragments (shared by both query tiles)
    const __bf16* kr0 = qkv + (size_t)(kb + lm) * LDQ + DD + h * DHH + half * 16;
    const __bf16* kr1 = qkv + (size_t)(kb + 16 + lm) * LDQ + DD + h * DHH + half * 16;
    bf16x16 b0 = *(const bf16x16*)(kr0);
    bf16x16 b1 = *(const bf16x16*)(kr0 + 32);
    bf16x16 b2 = *(const bf16x16*)(kr1);
    bf16x16 b3 = *(const bf16x16*)(kr1 + 32);

#pragma unroll
    for (int qt = 0; qt < 2; ++qt) {
      floatx8 s0 = zero8(), s1 = zero8();
      s0 = __builtin_amdgcn_wmma_f32_16x16x32_bf16(false, aq[qt][0], false, b0, (short)0, s0, false, false);
      s0 = __builtin_amdgcn_wmma_f32_16x16x32_bf16(false, aq[qt][1], false, b1, (short)0, s0, false, false);
      s1 = __builtin_amdgcn_wmma_f32_16x16x32_bf16(false, aq[qt][0], false, b2, (short)0, s1, false, false);
      s1 = __builtin_amdgcn_wmma_f32_16x16x32_bf16(false, aq[qt][1], false, b3, (short)0, s1, false, false);
      // exp + per-lane partial row sums; stage P in LDS (C-layout -> A-layout)
#pragma unroll
      for (int v = 0; v < 8; ++v) {
        float p0 = __expf(s0[v] * scale);
        float p1 = __expf(s1[v] * scale);
        Lpart[qt][v] += p0 + p1;
        int r = v + half * 8;
        pS[qt][r * 32 + lm]      = f2bf(p0);
        pS[qt][r * 32 + 16 + lm] = f2bf(p1);
      }
    }
    __syncthreads();
    bf16x16 pa[2];
#pragma unroll
    for (int qt = 0; qt < 2; ++qt) {
#pragma unroll
      for (int i = 0; i < 8; ++i) {
        pa[qt][i]     = pS[qt][lm * 32 + half * 16 + i];
        pa[qt][i + 8] = pS[qt][lm * 32 + half * 16 + 8 + i];
      }
    }
    __syncthreads();
    // ---- O += P @ V, V^T fragments shared by both query tiles
#pragma unroll
    for (int g = 0; g < 4; ++g) {
      bf16x16 bv = *(const bf16x16*)(vt + (size_t)(h * DHH + g * 16 + lm) * NN + kb + half * 16);
      oacc[0][g] = __builtin_amdgcn_wmma_f32_16x16x32_bf16(
          false, pa[0], false, bv, (short)0, oacc[0][g], false, false);
      oacc[1][g] = __builtin_amdgcn_wmma_f32_16x16x32_bf16(
          false, pa[1], false, bv, (short)0, oacc[1][g], false, false);
    }
  }
  // ---- reduce row sums across the 16 lanes of each half, normalize, write
#pragma unroll
  for (int qt = 0; qt < 2; ++qt) {
#pragma unroll
    for (int v = 0; v < 8; ++v) {
      float rs = Lpart[qt][v];
      rs += __shfl_xor(rs, 1, 16);
      rs += __shfl_xor(rs, 2, 16);
      rs += __shfl_xor(rs, 4, 16);
      rs += __shfl_xor(rs, 8, 16);
      float inv = 1.f / rs;
      int row = qb + qt * 16 + v + half * 8;
#pragma unroll
      for (int g = 0; g < 4; ++g) {
        int col = h * DHH + g * 16 + lm;
        o[(size_t)row * DD + col] = oacc[qt][g][v] * inv;
      }
    }
  }
}

// ---------------- BatchNorm stats: mean + invstd per column ----------------
__global__ void bn_stats_kernel(const float* __restrict__ x, float* __restrict__ st) {
  __shared__ float ss[256], sq[256];
  int col = blockIdx.x;
  float s = 0.f, q = 0.f;
  for (int r = threadIdx.x; r < NN; r += 256) {
    float v = x[(size_t)r * DD + col];
    s += v; q += v * v;
  }
  ss[threadIdx.x] = s; sq[threadIdx.x] = q;
  __syncthreads();
  for (int o = 128; o > 0; o >>= 1) {
    if (threadIdx.x < o) { ss[threadIdx.x] += ss[threadIdx.x + o];
                           sq[threadIdx.x] += sq[threadIdx.x + o]; }
    __syncthreads();
  }
  if (threadIdx.x == 0) {
    float mean = ss[0] / NN;
    float var  = sq[0] / NN - mean * mean;
    st[col]      = mean;
    st[DD + col] = rsqrtf(var + EPS);
  }
}

// ---------------- BN apply: out = (x-m)*istd*g + b (+ add) ----------------
__global__ void bn_apply_kernel(const float* __restrict__ x, const float* __restrict__ st,
                                const float* __restrict__ g, const float* __restrict__ b,
                                const float* __restrict__ add, float* __restrict__ out, int n) {
  int i = blockIdx.x * 256 + threadIdx.x;
  if (i >= n) return;
  int c = i & (DD - 1);
  float v = (x[i] - st[c]) * st[DD + c] * g[c] + b[c];
  out[i] = add ? v + add[i] : v;
}

// ---------------------------- host launcher ----------------------------
extern "C" void kernel_launch(void* const* d_in, const int* in_sizes, int n_in,
                              void* d_out, int out_size, void* d_ws, size_t ws_size,
                              hipStream_t stream) {
  const float* x       = (const float*)d_in[0];
  const int*   ei      = (const int*)  d_in[1];
  const float* ea      = (const float*)d_in[2];
  const float* w_gin1  = (const float*)d_in[3];
  const float* b_gin1  = (const float*)d_in[4];
  const float* w_gin2  = (const float*)d_in[5];
  const float* b_gin2  = (const float*)d_in[6];
  const float* w_qkv   = (const float*)d_in[7];
  const float* b_qkv   = (const float*)d_in[8];
  const float* w_o     = (const float*)d_in[9];
  const float* b_o     = (const float*)d_in[10];
  const float* bn1l_g  = (const float*)d_in[11];
  const float* bn1l_b  = (const float*)d_in[12];
  const float* bn1a_g  = (const float*)d_in[13];
  const float* bn1a_b  = (const float*)d_in[14];
  const float* w_ff1   = (const float*)d_in[15];
  const float* b_ff1   = (const float*)d_in[16];
  const float* w_ff2   = (const float*)d_in[17];
  const float* b_ff2   = (const float*)d_in[18];
  const float* bn2_g   = (const float*)d_in[19];
  const float* bn2_b   = (const float*)d_in[20];
  float* out = (float*)d_out;

  const size_t ND = (size_t)NN * DD;
  const size_t DSQ = (size_t)DD * DD;

  // f32 region
  float* B0 = (float*)d_ws;      // agg / gin-in, later h_attn pre-BN
  float* B1 = B0 + ND;           // gin hidden, later attention output o
  float* B2 = B1 + ND;           // h_local pre-BN, later FFN out pre-BN
  float* B3 = B2 + ND;           // h_local (post-BN)
  float* B4 = B3 + ND;           // qkv f32 [N,3D], later ff1 f32 [N,2D]
  float* B5 = B4 + 3 * ND;       // h = h_local + h_attn
  float* ST = B5 + ND;           // 2*D BN stats
  // bf16 region
  __bf16* WG1  = (__bf16*)(ST + 2 * DD);   // w_gin1        [D*D]
  __bf16* WG2  = WG1 + DSQ;                // w_gin2        [D*D]
  __bf16* WQKV = WG2 + DSQ;                // w_qkv         [3D*D]
  __bf16* WO   = WQKV + 3 * DSQ;           // w_o           [D*D]
  __bf16* WF1  = WO + DSQ;                 // w_ff1         [2D*D]
  __bf16* WF2  = WF1 + 2 * DSQ;            // w_ff2         [D*2D]
  __bf16* XB   = WF2 + 2 * DSQ;            // x  (bf16)     [N*D]
  __bf16* QKVB = XB + ND;                  // qkv (bf16)    [N*3D]
  __bf16* AB   = QKVB + 3 * ND;            // generic A operand scratch [N*2D]
  __bf16* VT   = AB + 2 * ND;              // V^T (bf16)    [D][N]

  const int ew = (int)(ND / 256);
  auto cvtBlocks = [](size_t n) { return (int)(n / 2048); };

  // 0) one-time conversions: weights + x
  cvt_bf16_kernel<<<cvtBlocks(DSQ),     256, 0, stream>>>(w_gin1, WG1,  (int)DSQ);
  cvt_bf16_kernel<<<cvtBlocks(DSQ),     256, 0, stream>>>(w_gin2, WG2,  (int)DSQ);
  cvt_bf16_kernel<<<cvtBlocks(3 * DSQ), 256, 0, stream>>>(w_qkv,  WQKV, (int)(3 * DSQ));
  cvt_bf16_kernel<<<cvtBlocks(DSQ),     256, 0, stream>>>(w_o,    WO,   (int)DSQ);
  cvt_bf16_kernel<<<cvtBlocks(2 * DSQ), 256, 0, stream>>>(w_ff1,  WF1,  (int)(2 * DSQ));
  cvt_bf16_kernel<<<cvtBlocks(2 * DSQ), 256, 0, stream>>>(w_ff2,  WF2,  (int)(2 * DSQ));
  cvt_bf16_kernel<<<cvtBlocks(ND),      256, 0, stream>>>(x,      XB,   (int)ND);

  // 1) GINE aggregation: B0 = x + sum_j relu(x_src + e)
  hipMemsetAsync(B0, 0, ND * sizeof(float), stream);
  edge_agg_kernel<<<EE, 128, 0, stream>>>(x, ei, ea, B0);
  add_inplace_kernel<<<ew, 256, 0, stream>>>(B0, x, (int)ND);

  // 2) GIN MLP
  cvt_bf16_kernel<<<cvtBlocks(ND), 256, 0, stream>>>(B0, AB, (int)ND);
  wmma_gemm_kernel<<<dim3(DD / 64, NN / 32), 32, 0, stream>>>(AB, WG1, b_gin1, nullptr, B1, NN, DD, DD, 1);
  cvt_bf16_kernel<<<cvtBlocks(ND), 256, 0, stream>>>(B1, AB, (int)ND);
  wmma_gemm_kernel<<<dim3(DD / 64, NN / 32), 32, 0, stream>>>(AB, WG2, b_gin2, x, B2, NN, DD, DD, 0);

  // 3) QKV projection + flash attention + output projection
  wmma_gemm_kernel<<<dim3(3 * DD / 64, NN / 32), 32, 0, stream>>>(XB, WQKV, b_qkv, nullptr, B4, NN, 3 * DD, DD, 0);
  cvt_bf16_kernel<<<cvtBlocks(3 * ND), 256, 0, stream>>>(B4, QKVB, (int)(3 * ND));
  v_transpose_kernel<<<DD, 256, 0, stream>>>(B4, VT);
  attn_flash_kernel<<<dim3(NN / 32, HH), 32, 0, stream>>>(QKVB, VT, B1);
  cvt_bf16_kernel<<<cvtBlocks(ND), 256, 0, stream>>>(B1, AB, (int)ND);
  wmma_gemm_kernel<<<dim3(DD / 64, NN / 32), 32, 0, stream>>>(AB, WO, b_o, x, B0, NN, DD, DD, 0);

  // 4) BatchNorms for the two branches, combine into h
  bn_stats_kernel<<<DD, 256, 0, stream>>>(B2, ST);
  bn_apply_kernel<<<ew, 256, 0, stream>>>(B2, ST, bn1l_g, bn1l_b, nullptr, B3, (int)ND);
  bn_stats_kernel<<<DD, 256, 0, stream>>>(B0, ST);
  bn_apply_kernel<<<ew, 256, 0, stream>>>(B0, ST, bn1a_g, bn1a_b, B3, B5, (int)ND);

  // 5) FFN + residual + final BN
  cvt_bf16_kernel<<<cvtBlocks(ND), 256, 0, stream>>>(B5, AB, (int)ND);
  wmma_gemm_kernel<<<dim3(2 * DD / 64, NN / 32), 32, 0, stream>>>(AB, WF1, b_ff1, nullptr, B4, NN, 2 * DD, DD, 1);
  cvt_bf16_kernel<<<cvtBlocks(2 * ND), 256, 0, stream>>>(B4, AB, (int)(2 * ND));
  wmma_gemm_kernel<<<dim3(DD / 64, NN / 32), 32, 0, stream>>>(AB, WF2, b_ff2, B5, B2, NN, DD, 2 * DD, 0);
  bn_stats_kernel<<<DD, 256, 0, stream>>>(B2, ST);
  bn_apply_kernel<<<ew, 256, 0, stream>>>(B2, ST, bn2_g, bn2_b, nullptr, out, (int)ND);
}
